// RNN_83313775608289
// MI455X (gfx1250) — matrix operand order repaired and checked by reading, not compiled, not run on hardware
//
#include <hip/hip_runtime.h>

// ---------------------------------------------------------------------------
// Fused 4-layer tanh RNN + FC head for MI455X (gfx1250, wave32, WMMA).
// One wave per 16-row batch tile runs the whole T=1024 scan through all
// 4 layers + FC. Hidden state lives in registers as bf16 WMMA A-fragments;
// ALL weight B-fragments live in registers (576 VGPRs, 1 wave/SIMD);
// LDS holds only a 2.3KB per-wave transpose buffer.
// ---------------------------------------------------------------------------

#define TSTEPS 1024
#define BATCH  512
#define DIN    52
#define ROWS   72                    // transpose-buffer row stride (bf16 elems)

typedef __attribute__((ext_vector_type(16))) __bf16        v16bf;
typedef __attribute__((ext_vector_type(8)))  float         v8f;
typedef __attribute__((ext_vector_type(4)))  unsigned int  v4u;

union BF16x16 { v16bf v; v4u u[2]; unsigned short s[16]; };

__device__ __forceinline__ unsigned short f2bf(float f) {
  union { __bf16 b; unsigned short u; } c; c.b = (__bf16)f; return c.u;
}

__device__ __forceinline__ float fast_tanh(float x) {
  // tanh(x) = 1 - 2/(exp(2x)+1); v_exp + v_rcp + fma, no precise-div expansion
  const float e = __expf(2.0f * x);
  const float r = __builtin_amdgcn_rcpf(1.0f + e);
  return __builtin_fmaf(-2.0f, r, 1.0f);
}

// B fragment (32x16 K-major tile) loaded straight from a global f32 row,
// converted to bf16 and packed. Lane L holds column N = row, K kbase+0..7 and kbase+16..23.
// f4-granular predicate handles the K=52 -> 64 zero padding of Wi0.
__device__ __forceinline__ v16bf load_fragB_global(const float* __restrict__ W,
                                                   int K, int row, int kbase) {
  BF16x16 r;
  #pragma unroll
  for (int q = 0; q < 4; ++q) {
    const int kb = kbase + 4 * q + ((q >> 1) << 3);
    float4 v;
    if (kb < K) v = *(const float4*)(W + (size_t)row * K + kb);
    else        v = make_float4(0.f, 0.f, 0.f, 0.f);
    r.s[4 * q + 0] = f2bf(v.x); r.s[4 * q + 1] = f2bf(v.y);
    r.s[4 * q + 2] = f2bf(v.z); r.s[4 * q + 3] = f2bf(v.w);
  }
  return r.v;
}

__global__ __launch_bounds__(32, 1)   // 1 wave / SIMD: allow full VGPR budget
void rnn4_fused_wmma(const float* __restrict__ x,
                     const float* __restrict__ Wih0, const float* __restrict__ Whh0,
                     const float* __restrict__ bih0, const float* __restrict__ bhh0,
                     const float* __restrict__ WihL, const float* __restrict__ WhhL,
                     const float* __restrict__ bihL, const float* __restrict__ bhhL,
                     const float* __restrict__ Wfc,  const float* __restrict__ bfc,
                     float* __restrict__ y)
{
  __shared__ unsigned short tbuf[16 * ROWS];   // per-wave transpose buffer
  const int lane = threadIdx.x & 31;
  const int lo   = lane & 15;        // column / row-in-tile id
  const int half = lane >> 4;        // lane half selects K sub-range
  const int b0   = blockIdx.x * 16;  // batch tile base

  // ---- preload ALL weight B-fragments into registers (one-time) ----------
  // mat 2l = Wi_l, mat 2l+1 = Wh_l, mat 8 = Wfc;  wB[m][Ntile j][Kchunk kk]
  v16bf wB[9][4][2];
  {
    const float* srcs[9] = { Wih0, Whh0,
                             WihL,        WhhL,
                             WihL + 4096, WhhL + 4096,
                             WihL + 8192, WhhL + 8192,
                             Wfc };
    #pragma unroll
    for (int m = 0; m < 9; ++m) {
      const int K = (m == 0) ? DIN : 64;       // Wi0 zero-padded 52 -> 64
      #pragma unroll
      for (int j = 0; j < 4; ++j)
        #pragma unroll
        for (int kk = 0; kk < 2; ++kk)
          wB[m][j][kk] =
              load_fragB_global(srcs[m], K, 16 * j + lo, 32 * kk + 8 * half);
    }
  }

  // ---- per-lane bias values: layer biases (bi+bh) and fc bias, per N-tile --
  float bias[5][4];
  #pragma unroll
  for (int j = 0; j < 4; ++j) {
    const int n = lo + 16 * j;
    bias[0][j] = bih0[n] + bhh0[n];
    #pragma unroll
    for (int l = 1; l < 4; ++l)
      bias[l][j] = bihL[(l - 1) * 64 + n] + bhhL[(l - 1) * 64 + n];
    bias[4][j] = bfc[n];
  }

  // ---- hidden state as bf16 A-fragments (zero init): h[layer][Kchunk] -----
  BF16x16 h[4][2];
  {
    v4u z = {0u, 0u, 0u, 0u};
    #pragma unroll
    for (int l = 0; l < 4; ++l)
      #pragma unroll
      for (int kk = 0; kk < 2; ++kk) { h[l][kk].u[0] = z; h[l][kk].u[1] = z; }
  }

  const float* xrow = x + (size_t)(b0 + lo) * TSTEPS * DIN;   // A-row m = lo

  for (int t = 0; t < TSTEPS; ++t) {
    // ---- layer-0 input: x_t tile as A-fragments, K padded 52 -> 64 -------
    // A layout per lane: row m = lane%16; elems 0..7 = K 8*half+0..7,
    // elems 8..15 = K 16+8*half+0..7 (within each 32-K chunk).
    BF16x16 xa[2];
    {
      const float* rp = xrow + t * DIN;
      #pragma unroll
      for (int kk = 0; kk < 2; ++kk) {
        #pragma unroll
        for (int seg = 0; seg < 2; ++seg) {
          const int kbase = 32 * kk + 16 * seg + 8 * half;
          #pragma unroll
          for (int q = 0; q < 2; ++q) {
            const int kb = kbase + 4 * q;
            float4 v;
            if (kb < DIN) v = *(const float4*)(rp + kb);   // exec-masked
            else          v = make_float4(0.f, 0.f, 0.f, 0.f);
            const int si = 8 * seg + 4 * q;
            xa[kk].s[si + 0] = f2bf(v.x); xa[kk].s[si + 1] = f2bf(v.y);
            xa[kk].s[si + 2] = f2bf(v.z); xa[kk].s[si + 3] = f2bf(v.w);
          }
        }
      }
    }

    // ---- 4 RNN layers: h_l = tanh(in @ Wi^T + bias + h_l @ Wh^T) ---------
    #pragma unroll
    for (int l = 0; l < 4; ++l) {
      const v16bf a0 = (l == 0) ? xa[0].v : h[l - 1][0].v;
      const v16bf a1 = (l == 0) ? xa[1].v : h[l - 1][1].v;

      #pragma unroll
      for (int j = 0; j < 4; ++j) {            // N-tiles of the 64 outputs
        v8f acc;
        #pragma unroll
        for (int e = 0; e < 8; ++e) acc[e] = bias[l][j];

        // input-GEMM part (critical chain)
        acc = __builtin_amdgcn_wmma_f32_16x16x32_bf16(
            false, a0, false, wB[2 * l][j][0], (short)0, acc, false, false);
        acc = __builtin_amdgcn_wmma_f32_16x16x32_bf16(
            false, a1, false, wB[2 * l][j][1], (short)0, acc, false, false);
        // recurrence part (uses prev-step h_l, off critical chain)
        acc = __builtin_amdgcn_wmma_f32_16x16x32_bf16(
            false, h[l][0].v, false, wB[2 * l + 1][j][0], (short)0, acc, false, false);
        acc = __builtin_amdgcn_wmma_f32_16x16x32_bf16(
            false, h[l][1].v, false, wB[2 * l + 1][j][1], (short)0, acc, false, false);

        // tanh + write C-layout tile into LDS transpose buffer as bf16
        #pragma unroll
        for (int e = 0; e < 8; ++e)
          tbuf[(e + 8 * half) * ROWS + lo + 16 * j] = f2bf(fast_tanh(acc[e]));
      }

      // reload as A-fragments for next layer / next time step
      // (DS ops are in-order per wave -> shared tbuf reuse is hazard-safe)
      #pragma unroll
      for (int kk = 0; kk < 2; ++kk) {
        const unsigned short* p = tbuf + lo * ROWS + 32 * kk + 8 * half;
        h[l][kk].u[0] = *(const v4u*)(p);
        h[l][kk].u[1] = *(const v4u*)(p + 16);
      }
    }

    // ---- FC head: y_t = h3 @ Wfc^T + bfc (linear) ------------------------
    #pragma unroll
    for (int j = 0; j < 4; ++j) {
      v8f acc;
      #pragma unroll
      for (int e = 0; e < 8; ++e) acc[e] = bias[4][j];
      acc = __builtin_amdgcn_wmma_f32_16x16x32_bf16(
          false, h[3][0].v, false, wB[8][j][0], (short)0, acc, false, false);
      acc = __builtin_amdgcn_wmma_f32_16x16x32_bf16(
          false, h[3][1].v, false, wB[8][j][1], (short)0, acc, false, false);
      // coalesced: lanes 0-15 write 64B contiguous per accumulator element
      #pragma unroll
      for (int e = 0; e < 8; ++e)
        y[((size_t)(b0 + e + 8 * half) * TSTEPS + t) * 64 + lo + 16 * j] = acc[e];
    }
  }
}

extern "C" void kernel_launch(void* const* d_in, const int* in_sizes, int n_in,
                              void* d_out, int out_size, void* d_ws, size_t ws_size,
                              hipStream_t stream) {
  const float* x    = (const float*)d_in[0];
  const float* Wih0 = (const float*)d_in[1];
  const float* Whh0 = (const float*)d_in[2];
  const float* bih0 = (const float*)d_in[3];
  const float* bhh0 = (const float*)d_in[4];
  const float* Wih  = (const float*)d_in[5];
  const float* Whh  = (const float*)d_in[6];
  const float* bih  = (const float*)d_in[7];
  const float* bhh  = (const float*)d_in[8];
  const float* Wfc  = (const float*)d_in[9];
  const float* bfc  = (const float*)d_in[10];
  float* y = (float*)d_out;

  (void)in_sizes; (void)n_in; (void)out_size; (void)d_ws; (void)ws_size;

  rnn4_fused_wmma<<<dim3(BATCH / 16), dim3(32), 0, stream>>>(
      x, Wih0, Whh0, bih0, bhh0, Wih, Whh, bih, bhh, Wfc, bfc, y);
}
